// GraphGPS_60146722013574
// MI455X (gfx1250) — matrix operand order repaired and checked by reading, not compiled, not run on hardware
//
#include <hip/hip_runtime.h>
#include <hip/hip_bf16.h>

// ---------------------------------------------------------------------------
// GraphGPS block for MI455X (gfx1250, wave32, WMMA bf16 16x16x32)
// - all dense GEMMs + both attention matmuls on v_wmma_f32_16x16x32_bf16
// - GEMM: 32x64 register tile per wave (8 WMMA / K-step, 0.75 loads per WMMA)
// - global attention: flash-style streaming, scores never materialized
// ---------------------------------------------------------------------------

#define D_    128
#define H_    4
#define C_    32
#define NA_   2048
#define NB_   1024
#define NC_   1024
#define NN_   4096
#define E_AB_ 65536
#define E_AC_ 32768
#define E_CB_ 32768
#define E_TOT_ (E_AB_ + E_AC_ + E_CB_ + NN_)   // 135168 (incl. self loops)
#define EPS_  1e-5f
#define SCALE_ 0.17677669529663687f            // 1/sqrt(32)

typedef __bf16 bf16_t;
typedef __attribute__((ext_vector_type(16))) __bf16 v16bf;
typedef __attribute__((ext_vector_type(8)))  __bf16 v8bf;
typedef __attribute__((ext_vector_type(8)))  float  v8f;

static __device__ inline v8bf ld8(const bf16_t* p) {
  return *reinterpret_cast<const v8bf*>(p);
}
static __device__ inline v16bf cat16(v8bf lo, v8bf hi) {
  return __builtin_shufflevector(lo, hi, 0,1,2,3,4,5,6,7,8,9,10,11,12,13,14,15);
}
static __device__ inline bf16_t tobf(float x) { return (bf16_t)x; }

// ---------------------------------------------------------------------------
// Elementwise / prep kernels
// ---------------------------------------------------------------------------
__global__ void k_fill(float* p, float v, int n) {
  int i = blockIdx.x * blockDim.x + threadIdx.x;
  if (i < n) p[i] = v;
}
__global__ void k_cvt(const float* __restrict__ in, bf16_t* __restrict__ out, int n) {
  int i = blockIdx.x * blockDim.x + threadIdx.x;
  if (i < n) out[i] = tobf(in[i]);
}
// transpose-convert: in is [R x Cc] f32 (row-major), out is [Cc x R] bf16
__global__ void k_cvt_t(const float* __restrict__ in, bf16_t* __restrict__ out,
                        int R, int Cc) {
  int i = blockIdx.x * blockDim.x + threadIdx.x;
  if (i < R * Cc) {
    int k = i / Cc, n = i % Cc;
    out[(size_t)n * R + k] = tobf(in[i]);
  }
}
__global__ void k_add(const float* __restrict__ a, const float* __restrict__ b,
                      float* __restrict__ o, int n) {
  int i = blockIdx.x * blockDim.x + threadIdx.x;
  if (i < n) o[i] = a[i] + b[i];
}

// ---------------------------------------------------------------------------
// bf16 WMMA GEMM: C[M x Nc] = A[M x K] * B[K x Nc] (+bias)(+res)(relu)
// A row-major bf16 (lda), Bt = B^T row-major bf16 (ldb = K stride).
// One wave computes a 32x64 tile: 2 A frags x 4 B frags -> 8 WMMAs per K-step.
// Requires M % 32 == 0, Nc % 64 == 0 (true for all call sites).
// ---------------------------------------------------------------------------
__global__ __launch_bounds__(128)
void k_gemm(const bf16_t* __restrict__ A, int lda,
            const bf16_t* __restrict__ Bt, int ldb,
            float* __restrict__ Cm,
            const float* __restrict__ bias,
            const float* __restrict__ res,
            int M, int Nc, int K, int relu) {
  const int lane = threadIdx.x & 31;
  const int wid  = blockIdx.x * 4 + (threadIdx.x >> 5);
  const int tilesN = Nc >> 6;
  const int tiles  = (M >> 5) * tilesN;
  if (wid >= tiles) return;                 // wave-uniform: EXEC stays all-ones
  const int tm = wid / tilesN, tn = wid % tilesN;
  const int lh   = lane & 15;
  const int aoff = (lane < 16) ? 0 : 16;    // A frag K-offset per lane half
  const int boff = (lane < 16) ? 0 : 16;    // B frag K-offset per lane half

  const bf16_t* arow0 = A + (size_t)(tm * 32 + lh) * lda;
  const bf16_t* arow1 = arow0 + (size_t)16 * lda;

  v8f acc[8];
#pragma unroll
  for (int j = 0; j < 8; ++j)
#pragma unroll
    for (int i = 0; i < 8; ++i) acc[j][i] = 0.f;

  for (int kk = 0; kk < K; kk += 32) {
    v16bf a0 = cat16(ld8(arow0 + kk + aoff), ld8(arow0 + kk + 8 + aoff));
    v16bf a1 = cat16(ld8(arow1 + kk + aoff), ld8(arow1 + kk + 8 + aoff));
#pragma unroll
    for (int j = 0; j < 4; ++j) {
      const bf16_t* bp = Bt + (size_t)(tn * 64 + j * 16 + lh) * ldb + boff + kk;
      v16bf bf = cat16(ld8(bp), ld8(bp + 8));
      acc[j]     = __builtin_amdgcn_wmma_f32_16x16x32_bf16(false, a0, false, bf,
                                                           (short)0, acc[j], false, false);
      acc[4 + j] = __builtin_amdgcn_wmma_f32_16x16x32_bf16(false, a1, false, bf,
                                                           (short)0, acc[4 + j], false, false);
    }
  }

  const int rb0 = tm * 32 + ((lane < 16) ? 0 : 8);
#pragma unroll
  for (int j = 0; j < 4; ++j) {
    const int col  = tn * 64 + j * 16 + lh;
    const float ba = bias ? bias[col] : 0.f;
#pragma unroll
    for (int r = 0; r < 8; ++r) {
      float v = acc[j][r] + ba;
      if (res)  v += res[(size_t)(rb0 + r) * Nc + col];
      if (relu) v = fmaxf(v, 0.f);
      Cm[(size_t)(rb0 + r) * Nc + col] = v;

      float w = acc[4 + j][r] + ba;
      if (res)  w += res[(size_t)(rb0 + 16 + r) * Nc + col];
      if (relu) w = fmaxf(w, 0.f);
      Cm[(size_t)(rb0 + 16 + r) * Nc + col] = w;
    }
  }
}

// ---------------------------------------------------------------------------
// Flash attention: one wave per (head, 16-query block); streams K/V in 32-key
// chunks with online softmax. Q,K read from bf16 qkv; V pre-transposed to
// Vt[h*32+c][n]. Scores never materialized (would be 256MB > L2).
// ---------------------------------------------------------------------------
__global__ __launch_bounds__(128)
void k_flash(const bf16_t* __restrict__ qkvb, const bf16_t* __restrict__ Vt,
             float* __restrict__ O) {
  __shared__ __align__(16) bf16_t P[4 * 16 * 32];   // per-wave P staging
  const int lane = threadIdx.x & 31;
  const int w    = threadIdx.x >> 5;
  const int wid  = blockIdx.x * 4 + w;
  const int hh   = wid >> 8;          // head 0..3
  const int qb   = wid & 255;         // query block 0..255
  const int lh   = lane & 15;
  const int aoff = (lane < 16) ? 0 : 16;
  const int boff = (lane < 16) ? 0 : 16;

  // Q fragment (A layout), head-sliced from qkv row
  const bf16_t* qrow = qkvb + (size_t)(qb * 16 + lh) * 384 + hh * 32;
  const v16bf qf = cat16(ld8(qrow + aoff), ld8(qrow + 8 + aoff));

  v8f o0, o1;
  float mrun[8], lrun[8];
#pragma unroll
  for (int r = 0; r < 8; ++r) { o0[r] = 0.f; o1[r] = 0.f; mrun[r] = -3.0e38f; lrun[r] = 0.f; }

  bf16_t* myP = &P[w * 512];

  for (int kk = 0; kk < NN_; kk += 32) {
    // K^T fragments: Bt row n(=key) is the key's bf16 row (contiguous c)
    const bf16_t* k0 = qkvb + (size_t)(kk +      lh) * 384 + 128 + hh * 32 + boff;
    const bf16_t* k1 = qkvb + (size_t)(kk + 16 + lh) * 384 + 128 + hh * 32 + boff;
    v16bf kf0 = cat16(ld8(k0), ld8(k0 + 8));
    v16bf kf1 = cat16(ld8(k1), ld8(k1 + 8));
    v8f z;
#pragma unroll
    for (int i = 0; i < 8; ++i) z[i] = 0.f;
    v8f s0 = __builtin_amdgcn_wmma_f32_16x16x32_bf16(false, qf, false, kf0,
                                                     (short)0, z, false, false);
    v8f s1 = __builtin_amdgcn_wmma_f32_16x16x32_bf16(false, qf, false, kf1,
                                                     (short)0, z, false, false);
    // online softmax per query row; rows live per lane-half, matching C layout
#pragma unroll
    for (int r = 0; r < 8; ++r) {
      float a0 = s0[r] * SCALE_, a1 = s1[r] * SCALE_;
      float cm = fmaxf(a0, a1);
#pragma unroll
      for (int off = 1; off < 16; off <<= 1) cm = fmaxf(cm, __shfl_xor(cm, off, 16));
      float nm   = fmaxf(mrun[r], cm);
      float corr = __expf(mrun[r] - nm);
      float p0   = __expf(a0 - nm);
      float p1   = __expf(a1 - nm);
      float ps   = p0 + p1;
#pragma unroll
      for (int off = 1; off < 16; off <<= 1) ps += __shfl_xor(ps, off, 16);
      lrun[r] = lrun[r] * corr + ps;
      mrun[r] = nm;
      o0[r] *= corr;
      o1[r] *= corr;
      // stage P (16x32) row-major in LDS for A-fragment reload
      int mloc = ((lane < 16) ? 0 : 8) + r;
      myP[mloc * 32 + lh]      = tobf(p0);
      myP[mloc * 32 + 16 + lh] = tobf(p1);
    }
    // P as A fragment (compiler inserts s_wait_dscnt for the LDS RAW)
    v16bf pf = cat16(ld8(myP + lh * 32 + aoff), ld8(myP + lh * 32 + 8 + aoff));
    // V fragments from Vt[h*32+c][n]: contiguous keys per lane
    const bf16_t* v0 = Vt + (size_t)(hh * 32 +      lh) * NN_ + kk + boff;
    const bf16_t* v1 = Vt + (size_t)(hh * 32 + 16 + lh) * NN_ + kk + boff;
    v16bf vf0 = cat16(ld8(v0), ld8(v0 + 8));
    v16bf vf1 = cat16(ld8(v1), ld8(v1 + 8));
    o0 = __builtin_amdgcn_wmma_f32_16x16x32_bf16(false, pf, false, vf0,
                                                 (short)0, o0, false, false);
    o1 = __builtin_amdgcn_wmma_f32_16x16x32_bf16(false, pf, false, vf1,
                                                 (short)0, o1, false, false);
  }

  const int rbase = qb * 16 + ((lane < 16) ? 0 : 8);
#pragma unroll
  for (int r = 0; r < 8; ++r) {
    float inv = 1.f / lrun[r];
    O[(size_t)(rbase + r) * D_ + hh * 32 + lh]      = o0[r] * inv;
    O[(size_t)(rbase + r) * D_ + hh * 32 + 16 + lh] = o1[r] * inv;
  }
}

// ---------------------------------------------------------------------------
// GAT helpers
// ---------------------------------------------------------------------------
__global__ void k_att(const float* __restrict__ xp,
                      const float* __restrict__ att_src,
                      const float* __restrict__ att_dst,
                      float* __restrict__ as, float* __restrict__ ad) {
  int i = blockIdx.x * blockDim.x + threadIdx.x;
  if (i >= NN_ * H_) return;
  int n = i >> 2, h = i & 3;
  float s = 0.f, d = 0.f;
  const float* row = xp + (size_t)n * D_ + h * C_;
#pragma unroll
  for (int c = 0; c < C_; ++c) { s += row[c] * att_src[h * C_ + c]; d += row[c] * att_dst[h * C_ + c]; }
  as[i] = s; ad[i] = d;
}

__global__ void k_edges(const int* ab_s, const int* ab_d, const int* ac_s,
                        const int* ac_d, const int* cb_s, const int* cb_d,
                        int* es, int* ed) {
  int i = blockIdx.x * blockDim.x + threadIdx.x;
  if (i >= E_TOT_) return;
  int s, d;
  if (i < E_AB_)                       { s = ab_s[i];                 d = ab_d[i] + NA_; }
  else if (i < E_AB_ + E_AC_)          { int j = i - E_AB_;           s = ac_s[j]; d = ac_d[j] + NA_ + NB_; }
  else if (i < E_AB_ + E_AC_ + E_CB_)  { int j = i - E_AB_ - E_AC_;   s = cb_s[j] + NA_ + NB_; d = cb_d[j] + NA_; }
  else                                 { int j = i - E_AB_ - E_AC_ - E_CB_; s = j; d = j; }
  es[i] = s; ed[i] = d;
}

static __device__ inline void atomicMaxF(float* a, float v) {
  unsigned* ua = (unsigned*)a;
  unsigned old = __float_as_uint(*a);
  while (__uint_as_float(old) < v) {
    unsigned cur = atomicCAS(ua, old, __float_as_uint(v));
    if (cur == old) break;
    old = cur;
  }
}

__global__ void k_e1(const int* __restrict__ es, const int* __restrict__ ed,
                     const float* __restrict__ as, const float* __restrict__ ad,
                     float* __restrict__ emax) {
  int i = blockIdx.x * blockDim.x + threadIdx.x;
  if (i >= E_TOT_) return;
  int s = es[i], d = ed[i];
#pragma unroll
  for (int h = 0; h < H_; ++h) {
    float e = as[s * H_ + h] + ad[d * H_ + h];
    e = (e > 0.f) ? e : 0.2f * e;
    atomicMaxF(&emax[d * H_ + h], e);
  }
}

__global__ void k_e2(const int* __restrict__ es, const int* __restrict__ ed,
                     const float* __restrict__ as, const float* __restrict__ ad,
                     const float* __restrict__ emax, float* __restrict__ eexp,
                     float* __restrict__ den) {
  int i = blockIdx.x * blockDim.x + threadIdx.x;
  if (i >= E_TOT_) return;
  int s = es[i], d = ed[i];
#pragma unroll
  for (int h = 0; h < H_; ++h) {
    float e = as[s * H_ + h] + ad[d * H_ + h];
    e = (e > 0.f) ? e : 0.2f * e;
    float ee = __expf(e - emax[d * H_ + h]);
    eexp[(size_t)i * H_ + h] = ee;
    atomicAdd(&den[d * H_ + h], ee);
  }
}

__global__ void k_e3(const int* __restrict__ es, const int* __restrict__ ed,
                     const float* __restrict__ eexp, const float* __restrict__ den,
                     const float* __restrict__ xp, float* __restrict__ hloc) {
  int i = blockIdx.x * blockDim.x + threadIdx.x;
  if (i >= E_TOT_ * H_) return;
  int e = i / H_, h = i % H_;
  int s = es[e], d = ed[e];
  float alpha = eexp[(size_t)e * H_ + h] / den[d * H_ + h];
  const float* src = xp + (size_t)s * D_ + h * C_;
  float* dst = hloc + (size_t)d * D_ + h * C_;
#pragma unroll
  for (int c = 0; c < C_; ++c) atomicAdd(&dst[c], src[c] * alpha);
}

// V transpose for flash: Vt[j][n] = qkv[n][256 + j], bf16
__global__ void k_vt(const float* __restrict__ qkv, bf16_t* __restrict__ Vt) {
  int i = blockIdx.x * blockDim.x + threadIdx.x;
  if (i >= NN_ * D_) return;
  int n = i >> 7, j = i & 127;
  Vt[(size_t)j * NN_ + n] = tobf(qkv[(size_t)n * 384 + 256 + j]);
}

// ---------------------------------------------------------------------------
// BatchNorm (biased var, as in reference)
// ---------------------------------------------------------------------------
__global__ void k_bnstats(const float* __restrict__ x, const float* __restrict__ res,
                          const float* __restrict__ cb, float* __restrict__ mu,
                          float* __restrict__ rs, int rows, int cols) {
  __shared__ float ss[256], sq[256];
  int j = blockIdx.x, t = threadIdx.x;
  float b = cb ? cb[j] : 0.f;
  float s = 0.f, q = 0.f;
  for (int i = t; i < rows; i += 256) {
    float v = x[(size_t)i * cols + j] + (res ? res[(size_t)i * cols + j] : 0.f) + b;
    s += v; q += v * v;
  }
  ss[t] = s; sq[t] = q; __syncthreads();
  for (int st = 128; st > 0; st >>= 1) {
    if (t < st) { ss[t] += ss[t + st]; sq[t] += sq[t + st]; }
    __syncthreads();
  }
  if (t == 0) {
    float m = ss[0] / rows;
    float var = sq[0] / rows - m * m;
    mu[j] = m; rs[j] = rsqrtf(var + EPS_);
  }
}

__global__ void k_bnapply(const float* __restrict__ x, const float* __restrict__ res,
                          const float* __restrict__ cb, const float* __restrict__ mu,
                          const float* __restrict__ rs, const float* __restrict__ g,
                          const float* __restrict__ be, float* __restrict__ out,
                          int rows, int cols) {
  int i = blockIdx.x * blockDim.x + threadIdx.x;
  if (i >= rows * cols) return;
  int j = i % cols;
  float v = x[i] + (res ? res[i] : 0.f) + (cb ? cb[j] : 0.f);
  out[i] = (v - mu[j]) * rs[j] * g[j] + be[j];
}

// ---------------------------------------------------------------------------
// Host launch
// ---------------------------------------------------------------------------
extern "C" void kernel_launch(void* const* d_in, const int* in_sizes, int n_in,
                              void* d_out, int out_size, void* d_ws, size_t ws_size,
                              hipStream_t stream) {
  (void)in_sizes; (void)n_in; (void)out_size; (void)ws_size;
  const float* x_a   = (const float*)d_in[0];
  const float* x_b   = (const float*)d_in[1];
  const float* x_c   = (const float*)d_in[2];
  const int* ab_s    = (const int*)d_in[3];
  const int* ab_d    = (const int*)d_in[4];
  const int* ac_s    = (const int*)d_in[5];
  const int* ac_d    = (const int*)d_in[6];
  const int* cb_s    = (const int*)d_in[7];
  const int* cb_d    = (const int*)d_in[8];
  const float* W_in_a = (const float*)d_in[9];
  const float* b_in_a = (const float*)d_in[10];
  const float* W_in_b = (const float*)d_in[11];
  const float* b_in_b = (const float*)d_in[12];
  const float* W_in_c = (const float*)d_in[13];
  const float* b_in_c = (const float*)d_in[14];
  const float* W_gat  = (const float*)d_in[15];
  const float* att_s  = (const float*)d_in[16];
  const float* att_d  = (const float*)d_in[17];
  const float* b_gat  = (const float*)d_in[18];
  const float* W_qkv  = (const float*)d_in[19];
  const float* b_qkv  = (const float*)d_in[20];
  const float* W_o    = (const float*)d_in[21];
  const float* b_o    = (const float*)d_in[22];
  const float* g1 = (const float*)d_in[23]; const float* be1 = (const float*)d_in[24];
  const float* g2 = (const float*)d_in[25]; const float* be2 = (const float*)d_in[26];
  const float* g3 = (const float*)d_in[27]; const float* be3 = (const float*)d_in[28];
  const float* W_mlp1 = (const float*)d_in[29];
  const float* b_mlp1 = (const float*)d_in[30];
  const float* W_mlp2 = (const float*)d_in[31];
  const float* b_mlp2 = (const float*)d_in[32];

  char* W = (char*)d_ws;
  size_t off = 0;
  auto ALLOC = [&](size_t bytes) -> void* {
    void* p = W + off;
    off = (off + bytes + 255) & ~(size_t)255;
    return p;
  };

  bf16_t* xb   = (bf16_t*)ALLOC((size_t)NN_ * D_ * 2);
  float*  h    = (float*) ALLOC((size_t)NN_ * D_ * 4);
  bf16_t* hb   = (bf16_t*)ALLOC((size_t)NN_ * D_ * 2);
  bf16_t* Wta  = (bf16_t*)ALLOC(D_ * D_ * 2);
  bf16_t* Wtb  = (bf16_t*)ALLOC(D_ * D_ * 2);
  bf16_t* Wtc  = (bf16_t*)ALLOC(D_ * D_ * 2);
  bf16_t* Wtg  = (bf16_t*)ALLOC(D_ * D_ * 2);
  bf16_t* Wqb  = (bf16_t*)ALLOC(3 * D_ * D_ * 2);
  bf16_t* Wob  = (bf16_t*)ALLOC(D_ * D_ * 2);
  bf16_t* Wt1  = (bf16_t*)ALLOC(2 * D_ * D_ * 2);
  bf16_t* Wt2  = (bf16_t*)ALLOC(2 * D_ * D_ * 2);
  float*  xp   = (float*) ALLOC((size_t)NN_ * D_ * 4);
  float*  as_  = (float*) ALLOC((size_t)NN_ * H_ * 4);
  float*  ad_  = (float*) ALLOC((size_t)NN_ * H_ * 4);
  int*    es   = (int*)   ALLOC((size_t)E_TOT_ * 4);
  int*    ed   = (int*)   ALLOC((size_t)E_TOT_ * 4);
  float*  emax = (float*) ALLOC((size_t)NN_ * H_ * 4);
  float*  den  = (float*) ALLOC((size_t)NN_ * H_ * 4);
  float*  eexp = (float*) ALLOC((size_t)E_TOT_ * H_ * 4);
  float*  hloc = (float*) ALLOC((size_t)NN_ * D_ * 4);
  float*  h1   = (float*) ALLOC((size_t)NN_ * D_ * 4);
  float*  qkv  = (float*) ALLOC((size_t)NN_ * 3 * D_ * 4);
  bf16_t* qkvb = (bf16_t*)ALLOC((size_t)NN_ * 3 * D_ * 2);
  bf16_t* Vt   = (bf16_t*)ALLOC((size_t)D_ * NN_ * 2);
  float*  o    = (float*) ALLOC((size_t)NN_ * D_ * 4);
  bf16_t* ob   = (bf16_t*)ALLOC((size_t)NN_ * D_ * 2);
  float*  opj  = (float*) ALLOC((size_t)NN_ * D_ * 4);
  float*  h2   = (float*) ALLOC((size_t)NN_ * D_ * 4);
  float*  osum = (float*) ALLOC((size_t)NN_ * D_ * 4);
  bf16_t* outb = (bf16_t*)ALLOC((size_t)NN_ * D_ * 2);
  float*  hid  = (float*) ALLOC((size_t)NN_ * 2 * D_ * 4);
  bf16_t* hidb = (bf16_t*)ALLOC((size_t)NN_ * 2 * D_ * 2);
  float*  out2 = (float*) ALLOC((size_t)NN_ * D_ * 4);
  float*  mu   = (float*) ALLOC(D_ * 4);
  float*  rs   = (float*) ALLOC(D_ * 4);

  auto GB = [](int n) { return (unsigned)((n + 255) / 256); };
  auto GT = [](int M, int Nc) {                 // GEMM grid: 32x64 tiles, 4 waves/block
    int tiles = (M / 32) * (Nc / 64);
    return (unsigned)((tiles + 3) / 4);
  };

  // ---- prep: convert inputs and weights to bf16 (weights pre-transposed) ---
  k_cvt<<<GB(NA_ * D_), 256, 0, stream>>>(x_a, xb, NA_ * D_);
  k_cvt<<<GB(NB_ * D_), 256, 0, stream>>>(x_b, xb + (size_t)NA_ * D_, NB_ * D_);
  k_cvt<<<GB(NC_ * D_), 256, 0, stream>>>(x_c, xb + (size_t)(NA_ + NB_) * D_, NC_ * D_);
  k_cvt_t<<<GB(D_ * D_), 256, 0, stream>>>(W_in_a, Wta, D_, D_);
  k_cvt_t<<<GB(D_ * D_), 256, 0, stream>>>(W_in_b, Wtb, D_, D_);
  k_cvt_t<<<GB(D_ * D_), 256, 0, stream>>>(W_in_c, Wtc, D_, D_);
  k_cvt_t<<<GB(D_ * D_), 256, 0, stream>>>(W_gat, Wtg, D_, D_);
  k_cvt<<<GB(3 * D_ * D_), 256, 0, stream>>>(W_qkv, Wqb, 3 * D_ * D_);   // already B^T layout
  k_cvt<<<GB(D_ * D_), 256, 0, stream>>>(W_o, Wob, D_ * D_);             // already B^T layout
  k_cvt_t<<<GB(2 * D_ * D_), 256, 0, stream>>>(W_mlp1, Wt1, D_, 2 * D_);
  k_cvt_t<<<GB(2 * D_ * D_), 256, 0, stream>>>(W_mlp2, Wt2, 2 * D_, D_);

  // ---- per-type input projections -> h [N, D] ------------------------------
  k_gemm<<<GT(NA_, D_), 128, 0, stream>>>(xb, D_, Wta, D_, h, b_in_a, nullptr, NA_, D_, D_, 0);
  k_gemm<<<GT(NB_, D_), 128, 0, stream>>>(xb + (size_t)NA_ * D_, D_, Wtb, D_, h + (size_t)NA_ * D_, b_in_b, nullptr, NB_, D_, D_, 0);
  k_gemm<<<GT(NC_, D_), 128, 0, stream>>>(xb + (size_t)(NA_ + NB_) * D_, D_, Wtc, D_, h + (size_t)(NA_ + NB_) * D_, b_in_c, nullptr, NC_, D_, D_, 0);
  k_cvt<<<GB(NN_ * D_), 256, 0, stream>>>(h, hb, NN_ * D_);

  // ---- GAT message passing -------------------------------------------------
  k_gemm<<<GT(NN_, D_), 128, 0, stream>>>(hb, D_, Wtg, D_, xp, nullptr, nullptr, NN_, D_, D_, 0);
  k_att<<<GB(NN_ * H_), 256, 0, stream>>>(xp, att_s, att_d, as_, ad_);
  k_edges<<<GB(E_TOT_), 256, 0, stream>>>(ab_s, ab_d, ac_s, ac_d, cb_s, cb_d, es, ed);
  k_fill<<<GB(NN_ * D_), 256, 0, stream>>>(hloc, 0.f, NN_ * D_);
  k_fill<<<GB(NN_ * H_), 256, 0, stream>>>(emax, -3.0e38f, NN_ * H_);
  k_fill<<<GB(NN_ * H_), 256, 0, stream>>>(den, 0.f, NN_ * H_);
  k_e1<<<GB(E_TOT_), 256, 0, stream>>>(es, ed, as_, ad_, emax);
  k_e2<<<GB(E_TOT_), 256, 0, stream>>>(es, ed, as_, ad_, emax, eexp, den);
  k_e3<<<GB(E_TOT_ * H_), 256, 0, stream>>>(es, ed, eexp, den, xp, hloc);
  k_bnstats<<<D_, 256, 0, stream>>>(hloc, h, b_gat, mu, rs, NN_, D_);
  k_bnapply<<<GB(NN_ * D_), 256, 0, stream>>>(hloc, h, b_gat, mu, rs, g1, be1, h1, NN_, D_);

  // ---- global multi-head self-attention (flash) ----------------------------
  k_gemm<<<GT(NN_, 3 * D_), 128, 0, stream>>>(hb, D_, Wqb, D_, qkv, b_qkv, nullptr, NN_, 3 * D_, D_, 0);
  k_cvt<<<GB(NN_ * 3 * D_), 256, 0, stream>>>(qkv, qkvb, NN_ * 3 * D_);
  k_vt<<<GB(NN_ * D_), 256, 0, stream>>>(qkv, Vt);
  k_flash<<<256, 128, 0, stream>>>(qkvb, Vt, o);
  k_cvt<<<GB(NN_ * D_), 256, 0, stream>>>(o, ob, NN_ * D_);
  k_gemm<<<GT(NN_, D_), 128, 0, stream>>>(ob, D_, Wob, D_, opj, b_o, nullptr, NN_, D_, D_, 0);
  k_bnstats<<<D_, 256, 0, stream>>>(opj, h, nullptr, mu, rs, NN_, D_);
  k_bnapply<<<GB(NN_ * D_), 256, 0, stream>>>(opj, h, nullptr, mu, rs, g2, be2, h2, NN_, D_);

  // ---- combine + MLP -------------------------------------------------------
  k_add<<<GB(NN_ * D_), 256, 0, stream>>>(h1, h2, osum, NN_ * D_);
  k_cvt<<<GB(NN_ * D_), 256, 0, stream>>>(osum, outb, NN_ * D_);
  k_gemm<<<GT(NN_, 2 * D_), 128, 0, stream>>>(outb, D_, Wt1, D_, hid, b_mlp1, nullptr, NN_, 2 * D_, D_, 1);
  k_cvt<<<GB(NN_ * 2 * D_), 256, 0, stream>>>(hid, hidb, NN_ * 2 * D_);
  k_gemm<<<GT(NN_, D_), 128, 0, stream>>>(hidb, 2 * D_, Wt2, 2 * D_, out2, b_mlp2, osum, NN_, D_, 2 * D_, 0);

  // ---- final BN -> d_out (rows already in a,b,c order) ---------------------
  k_bnstats<<<D_, 256, 0, stream>>>(out2, nullptr, nullptr, mu, rs, NN_, D_);
  k_bnapply<<<GB(NN_ * D_), 256, 0, stream>>>(out2, nullptr, nullptr, mu, rs, g3, be3,
                                              (float*)d_out, NN_, D_);
}
